// BaselinePointerNetwork_31404800868581
// MI455X (gfx1250) — compile-verified, matrix operand order checked by reading, not run on hardware
//
#include <hip/hip_runtime.h>
#include <hip/hip_bf16.h>

// CVRP pointer-network for MI455X (gfx1250, wave32, WMMA).
// prep (transpose+f16 weights) -> encoder (WMMA attention + fused Wp1 split
// GEMMs) -> persistent decoder (async-to-LDS staged tiles, incremental ctx,
// wave-level reductions).

typedef __attribute__((ext_vector_type(16))) _Float16 v16h;
typedef __attribute__((ext_vector_type(8)))  _Float16 v8h;
typedef __attribute__((ext_vector_type(8)))  float    v8f;
typedef __attribute__((address_space(3)))    _Float16 lds_half;

#define BATCH 1024
#define NN    128
#define HH    128
#define TT    256
#define NEGV  (-1e9f)

// ---------------------------------------------------------------- WMMA helper
__device__ __forceinline__ v8f wmma16(v16h a, v16h b, v8f c) {
  return __builtin_amdgcn_wmma_f32_16x16x32_f16(
      /*neg_a=*/false, a, /*neg_b=*/false, b,
      /*c_mod=*/(short)0, c, /*reuse_a=*/false, /*reuse_b=*/false);
}

// Load a 16x32 f16 fragment (A-layout per ISA 7.12.2) from a row-major matrix.
__device__ __forceinline__ v16h frag_ld(const _Float16* base, int row0, int ldh,
                                        int kc, int lane) {
  int m = lane & 15;
  int khalf = (lane >> 4) & 1;               // lanes 16-31 carry K+8 groups
  const _Float16* p = base + (size_t)(row0 + m) * ldh + kc * 32 + khalf * 8;
  v8h lo = *(const v8h*)p;                   // K = kbase .. kbase+7
  v8h hi = *(const v8h*)(p + 16);            // K = 16+kbase .. 16+kbase+7
  v16h r;
#pragma unroll
  for (int i = 0; i < 8; i++) { r[i] = lo[i]; r[i + 8] = hi[i]; }
  return r;
}

// ------------------------------------------------------------------- prep
__global__ __launch_bounds__(256) void cvrp_prep(const float* __restrict__ Wq,
                                                 const float* __restrict__ Wk,
                                                 const float* __restrict__ Wv,
                                                 const float* __restrict__ Wp1,
                                                 _Float16* __restrict__ wt) {
  int idx = blockIdx.x * 256 + threadIdx.x;
  if (idx >= 5 * HH * HH) return;
  int mat = idx >> 14;
  int n = (idx >> 7) & 127;
  int k = idx & 127;
  float v;
  if (mat == 0)      v = Wq[k * HH + n];
  else if (mat == 1) v = Wk[k * HH + n];
  else if (mat == 2) v = Wv[k * HH + n];
  else if (mat == 3) v = Wp1[k * HH + n];              // top half rows 0..127
  else               v = Wp1[(k + HH) * HH + n];       // bottom rows 128..255
  wt[idx] = (_Float16)v;
}

// ------------------------------------------------------------------- encoder
#define ENC_SMEM (3 * 128 * 128 * 2 + 8 * 16 * 128 * 4 + 8 * 16 * 128 * 2 + 128 * 4)

__global__ __launch_bounds__(256) void cvrp_encoder(
    const float* __restrict__ coords, const float* __restrict__ demands,
    const float* __restrict__ W_emb, const float* __restrict__ b_emb,
    const float* __restrict__ bq, const float* __restrict__ bk,
    const float* __restrict__ bv, const float* __restrict__ bp1,
    const _Float16* __restrict__ WqT, const _Float16* __restrict__ WkT,
    const _Float16* __restrict__ WvT, const _Float16* __restrict__ W1aT,
    const _Float16* __restrict__ W1bT,
    _Float16* __restrict__ attF, _Float16* __restrict__ attB) {
  extern __shared__ char smem[];
  _Float16* embh = (_Float16*)smem;              // 32KB (later Q)
  _Float16* kh   = embh + 128 * 128;             // 32KB (later att_emb)
  _Float16* vT   = kh + 128 * 128;               // 32KB  V^T [h][node]
  float*    scr  = (float*)(vT + 128 * 128);     // 64KB per-wave scratch
  _Float16* prb  = (_Float16*)(scr + 8 * 16 * 128); // 32KB probs
  float*    rsum = (float*)(prb + 8 * 16 * 128); // 512B

  const int b = blockIdx.x;
  const int tid = threadIdx.x;
  const int lane = tid & 31;
  const int w = tid >> 5;
  const int rowbase = w * 16;
  const int col16 = lane & 15;
  const int hi = (lane >> 4) & 1;

  // ---- Phase 0: node features -> f16 embedding in LDS
  {
    int n = tid >> 1;
    int h0 = (tid & 1) * 64;
    float x = coords[((size_t)b * NN + n) * 2 + 0];
    float y = coords[((size_t)b * NN + n) * 2 + 1];
    float dm = demands[(size_t)b * NN + n];
    for (int h = 0; h < 64; h++) {
      int hc = h0 + h;
      float e = x * W_emb[hc] + y * W_emb[HH + hc] + dm * W_emb[2 * HH + hc] +
                b_emb[hc];
      embh[n * HH + hc] = (_Float16)e;
    }
  }
  __syncthreads();

  // ---- Phase 1: K, V, Q projections (WMMA). Each wave owns 16 rows.
  for (int ct = 0; ct < 8; ct++) {               // K -> kh (row-major)
    int col = ct * 16 + col16;
    v8f acc;  float bb = bk[col];
#pragma unroll
    for (int i = 0; i < 8; i++) acc[i] = bb;
#pragma unroll
    for (int kc = 0; kc < 4; kc++)
      acc = wmma16(frag_ld(embh, rowbase, HH, kc, lane),
                   frag_ld(WkT, ct * 16, HH, kc, lane), acc);
#pragma unroll
    for (int r = 0; r < 8; r++)
      kh[(rowbase + r + 8 * hi) * HH + col] = (_Float16)acc[r];
  }
  for (int ct = 0; ct < 8; ct++) {               // V -> vT (transposed)
    int col = ct * 16 + col16;
    v8f acc;  float bb = bv[col];
#pragma unroll
    for (int i = 0; i < 8; i++) acc[i] = bb;
#pragma unroll
    for (int kc = 0; kc < 4; kc++)
      acc = wmma16(frag_ld(embh, rowbase, HH, kc, lane),
                   frag_ld(WvT, ct * 16, HH, kc, lane), acc);
    v8h tmp;
#pragma unroll
    for (int r = 0; r < 8; r++) tmp[r] = (_Float16)acc[r];
    *(v8h*)&vT[col * HH + rowbase + 8 * hi] = tmp;
  }
  {                                              // Q -> overwrite own embh rows
    v8f qac[8];
#pragma unroll
    for (int ct = 0; ct < 8; ct++) {
      float bb = bq[ct * 16 + col16];
#pragma unroll
      for (int i = 0; i < 8; i++) qac[ct][i] = bb;
    }
    for (int ct = 0; ct < 8; ct++)
#pragma unroll
      for (int kc = 0; kc < 4; kc++)
        qac[ct] = wmma16(frag_ld(embh, rowbase, HH, kc, lane),
                         frag_ld(WqT, ct * 16, HH, kc, lane), qac[ct]);
    for (int ct = 0; ct < 8; ct++)
#pragma unroll
      for (int r = 0; r < 8; r++)
        embh[(rowbase + r + 8 * hi) * HH + ct * 16 + col16] =
            (_Float16)qac[ct][r];
  }
  __syncthreads();

  // ---- Phase 2: scores = Q@K^T / sqrt(H), per-wave row softmax
  const float sscale = 0.08838834764831843f;     // 1/sqrt(128)
  float* sw = scr + w * 16 * 128;
  _Float16* pw = prb + w * 16 * 128;
  for (int ct = 0; ct < 8; ct++) {
    v8f acc;
#pragma unroll
    for (int i = 0; i < 8; i++) acc[i] = 0.0f;
#pragma unroll
    for (int kc = 0; kc < 4; kc++)
      acc = wmma16(frag_ld(embh, rowbase, HH, kc, lane),     // Q rows
                   frag_ld(kh, ct * 16, HH, kc, lane), acc); // K rows as B^T
#pragma unroll
    for (int r = 0; r < 8; r++)
      sw[(r + 8 * hi) * 128 + ct * 16 + col16] = acc[r] * sscale;
  }
  if (lane < 16) {                               // lane = local row
    float* srow = sw + lane * 128;
    float mx = srow[0];
    for (int c = 1; c < 128; c++) mx = fmaxf(mx, srow[c]);
    float s = 0.0f;
    for (int c = 0; c < 128; c++) {
      float e = __expf(srow[c] - mx);
      pw[lane * 128 + c] = (_Float16)e;          // unnormalized probs
      s += e;
    }
    rsum[rowbase + lane] = s;
  }
  __syncthreads();                               // all waves done reading kh

  // ---- Phase 3: att_emb = (P@V)/rowsum -> overwrite own kh rows (f16)
  for (int ct = 0; ct < 8; ct++) {
    v8f acc;
#pragma unroll
    for (int i = 0; i < 8; i++) acc[i] = 0.0f;
#pragma unroll
    for (int kc = 0; kc < 4; kc++)
      acc = wmma16(frag_ld(pw, 0, 128, kc, lane),
                   frag_ld(vT, ct * 16, HH, kc, lane), acc);
#pragma unroll
    for (int r = 0; r < 8; r++) {
      float inv = 1.0f / rsum[rowbase + r + 8 * hi];
      kh[(rowbase + r + 8 * hi) * HH + ct * 16 + col16] =
          (_Float16)(acc[r] * inv);
    }
  }

  // ---- Phase 4: att_fixed = att_emb@Wp1_top + bp1 ; att_bot = att_emb@Wp1_bot
  _Float16* swh = (_Float16*)sw;                 // reuse scratch as f16 staging
  for (int pass = 0; pass < 2; pass++) {
    const _Float16* WT = pass ? W1bT : W1aT;
    _Float16* outp = pass ? attB : attF;
    for (int ct = 0; ct < 8; ct++) {
      int col = ct * 16 + col16;
      v8f acc;  float bb = pass ? 0.0f : bp1[col];
#pragma unroll
      for (int i = 0; i < 8; i++) acc[i] = bb;
#pragma unroll
      for (int kc = 0; kc < 4; kc++)
        acc = wmma16(frag_ld(kh, rowbase, HH, kc, lane),
                     frag_ld(WT, ct * 16, HH, kc, lane), acc);
#pragma unroll
      for (int r = 0; r < 8; r++)
        swh[(r + 8 * hi) * 128 + col] = (_Float16)acc[r];
    }
    size_t gbase = ((size_t)b * NN + rowbase) * HH + (size_t)lane * 64;
    const v8h* src = (const v8h*)(swh + lane * 64);
    v8h* dst = (v8h*)(outp + gbase);
#pragma unroll
    for (int i = 0; i < 8; i++) dst[i] = src[i];
  }
}

// ------------------------------------------------------------------- decoder
__device__ __forceinline__ float rng01(unsigned int b, unsigned int t) {
  unsigned int x = b * 747796405u + t * 2891336453u + 0x9E3779B9u;
  x ^= x >> 16; x *= 0x7feb352du;
  x ^= x >> 15; x *= 0x846ca68bu;
  x ^= x >> 16;
  return (float)(x >> 8) * (1.0f / 16777216.0f);
}

struct DecState {
  float rem_cap, lp_acc, mx;
  int at_depot, done, cnt, newvis;
  int pad;
};

// dynamic LDS: attf 32KB | attb 32KB | 6 float arrays | 2 int arrays | state
#define DEC_SMEM (2 * NN * HH * 2 + 6 * 128 * 4 + 2 * 128 * 4 + (int)sizeof(DecState))

__global__ __launch_bounds__(128) void cvrp_decoder(
    const float* __restrict__ demands, const float* __restrict__ caps,
    const float* __restrict__ Wp2, const float* __restrict__ bp2,
    const _Float16* __restrict__ attF, const _Float16* __restrict__ attB,
    float* __restrict__ out) {
  extern __shared__ char smem[];
  _Float16* attf   = (_Float16*)smem;            // [128][128] f16
  _Float16* attb   = attf + NN * HH;             // [128][128] f16
  float* S         = (float*)(attb + NN * HH);
  float* ctxp      = S + 128;
  float* dem       = ctxp + 128;
  float* w2        = dem + 128;
  float* sc        = w2 + 128;                   // logits
  float* ev        = sc + 128;                   // exp(logit - mx)
  int* maskb       = (int*)(ev + 128);
  int* visited     = maskb + 128;
  DecState* st     = (DecState*)(visited + 128);

  const int b = blockIdx.x;
  const int n = threadIdx.x;                     // node index / h index

  // ---- async DMA the two 32KB tiles straight into LDS (ASYNCcnt-tracked)
  {
    unsigned lf = (unsigned)(size_t)(lds_half*)attf;
    unsigned lb = (unsigned)(size_t)(lds_half*)attb;
    const _Float16* gf = attF + ((size_t)b * NN + n) * HH;
    const _Float16* gb = attB + ((size_t)b * NN + n) * HH;
#pragma unroll
    for (int i = 0; i < 16; i++) {
      unsigned la = lf + (unsigned)(n * 256 + i * 16);
      unsigned long long ga = (unsigned long long)(gf + i * 8);
      asm volatile("global_load_async_to_lds_b128 %0, %1, off"
                   :: "v"(la), "v"(ga) : "memory");
    }
#pragma unroll
    for (int i = 0; i < 16; i++) {
      unsigned la = lb + (unsigned)(n * 256 + i * 16);
      unsigned long long ga = (unsigned long long)(gb + i * 8);
      asm volatile("global_load_async_to_lds_b128 %0, %1, off"
                   :: "v"(la), "v"(ga) : "memory");
    }
  }
  dem[n] = demands[(size_t)b * NN + n];
  w2[n] = Wp2[n];
  visited[n] = 0;
  if (n == 0) {
    st->rem_cap = caps[b]; st->lp_acc = 0.0f;
    st->at_depot = 1; st->done = 0; st->cnt = NN; st->newvis = -1;
  }
#if __has_builtin(__builtin_amdgcn_s_wait_asynccnt)
  __builtin_amdgcn_s_wait_asynccnt(0);
#else
  asm volatile("s_wait_asynccnt 0x0" ::: "memory");
#endif
  __syncthreads();

  {                                              // S = sum_j att_bot[j][h]
    float s = 0.0f;
    for (int j = 0; j < NN; j++) s += (float)attb[j * HH + n];
    S[n] = s;
  }
  const float bias2 = bp2[0];
  __syncthreads();

  for (int t = 0; t < TT; t++) {
    // (1) context projection (S is already @Wp1_bot; depot never visited)
    ctxp[n] = S[n] / fmaxf((float)st->cnt, 1.0f);
    __syncthreads();
    // (2) per-node score: relu(att_fixed + ctxp) . Wp2 + bp2
    {
      float acc = bias2;
      float rc = st->rem_cap;
      const _Float16* arow = attf + n * HH;
#pragma unroll 8
      for (int h = 0; h < HH; h++)
        acc += fmaxf((float)arow[h] + ctxp[h], 0.0f) * w2[h];
      sc[n] = acc;
      maskb[n] = (visited[n] || (dem[n] > rc)) ? 1 : 0;
    }
    __syncthreads();
    // (3) wave-0 AND-reductions (all-visited, all-masked) + mask fixups
    if (n < 32) {
      int ad = st->at_depot;
      int a1 = 1, a2 = 1;
#pragma unroll
      for (int j = 0; j < 4; j++) {
        int idx = n + j * 32;
        a1 &= (idx == 0) ? 1 : visited[idx];
        int mb = maskb[idx];
        if (idx == 0) mb |= ad;
        a2 &= mb;
      }
      for (int off = 16; off; off >>= 1) {
        a1 &= __shfl_xor(a1, off, 32);
        a2 &= __shfl_xor(a2, off, 32);
      }
      if (n == 0) {
        int d = st->done | (a1 & ad);
        int m0 = maskb[0] | ad;
        if (a2 && !ad) m0 = 0;                   // allow return to depot
        d |= (a2 & ad);
        if (d) m0 = 0;
        maskb[0] = m0;
        st->done = d;
      }
    }
    __syncthreads();
    // (4) logits + max reduce (wave 0)
    sc[n] = maskb[n] ? NEGV : sc[n];
    __syncthreads();
    if (n < 32) {
      float m = -3e30f;
#pragma unroll
      for (int j = 0; j < 4; j++) m = fmaxf(m, sc[n + j * 32]);
      for (int off = 16; off; off >>= 1) m = fmaxf(m, __shfl_xor(m, off, 32));
      if (n == 0) st->mx = m;
    }
    __syncthreads();
    float mx = st->mx;
    ev[n] = __expf(sc[n] - mx);
    __syncthreads();
    // (5) sum reduce (wave 0) + serial CDF sampling + state update (lane 0)
    if (n < 32) {
      float s = 0.0f;
#pragma unroll
      for (int j = 0; j < 4; j++) s += ev[n + j * 32];
      for (int off = 16; off; off >>= 1) s += __shfl_xor(s, off, 32);
      if (n == 0) {
        float sum = s;
        float r = rng01((unsigned)b, (unsigned)t) * sum;
        float cum = 0.0f;
        int a = -1;
        for (int j = 0; j < NN; j++) {
          cum += ev[j];
          if (a < 0 && cum >= r) a = j;
        }
        if (a < 0) a = 0;
        float lp = sc[a] - mx - __logf(sum);
        int d = st->done;
        if (d) { a = 0; lp = 0.0f; }
        st->lp_acc += lp;
        int isdep = (a == 0);
        int upd = !d;
        int nv = (upd && !isdep) ? a : -1;
        if (nv > 0 && !visited[nv]) { visited[nv] = 1; st->cnt = st->cnt - 1; }
        if (upd) {
          st->rem_cap = isdep ? caps[b] : (st->rem_cap - dem[a]);
          st->at_depot = isdep;
        }
        st->newvis = nv;
        out[(size_t)b * TT + t] = (float)a;      // actions [B,T]
      }
    }
    __syncthreads();
    // (6) incremental context-sum update from LDS copy of att_bot
    {
      int nv = st->newvis;
      if (nv >= 0) S[n] -= (float)attb[nv * HH + n];
    }
    __syncthreads();
  }
  if (n == 0) out[(size_t)BATCH * TT + b] = st->lp_acc;  // log-probs [B]
}

// ------------------------------------------------------------------- launch
extern "C" void kernel_launch(void* const* d_in, const int* in_sizes, int n_in,
                              void* d_out, int out_size, void* d_ws,
                              size_t ws_size, hipStream_t stream) {
  const float* coords  = (const float*)d_in[0];
  const float* demands = (const float*)d_in[1];
  const float* caps    = (const float*)d_in[2];
  const float* W_emb   = (const float*)d_in[3];
  const float* b_emb   = (const float*)d_in[4];
  const float* Wq      = (const float*)d_in[5];
  const float* bq      = (const float*)d_in[6];
  const float* Wk      = (const float*)d_in[7];
  const float* bk      = (const float*)d_in[8];
  const float* Wv      = (const float*)d_in[9];
  const float* bv      = (const float*)d_in[10];
  const float* Wp1     = (const float*)d_in[11];
  const float* bp1     = (const float*)d_in[12];
  const float* Wp2     = (const float*)d_in[13];
  const float* bp2     = (const float*)d_in[14];
  float* out = (float*)d_out;

  _Float16* wt = (_Float16*)d_ws;                       // 5 x 128 x 128 f16
  _Float16* WqT  = wt;
  _Float16* WkT  = wt + 1 * HH * HH;
  _Float16* WvT  = wt + 2 * HH * HH;
  _Float16* W1aT = wt + 3 * HH * HH;
  _Float16* W1bT = wt + 4 * HH * HH;
  _Float16* attF = (_Float16*)((char*)d_ws + (1 << 18));          // 32MB
  _Float16* attB = attF + (size_t)BATCH * NN * HH;                // 32MB

  cvrp_prep<<<(5 * HH * HH + 255) / 256, 256, 0, stream>>>(Wq, Wk, Wv, Wp1, wt);

  cvrp_encoder<<<BATCH, 256, ENC_SMEM, stream>>>(
      coords, demands, W_emb, b_emb, bq, bk, bv, bp1,
      WqT, WkT, WvT, W1aT, W1bT, attF, attB);

  cvrp_decoder<<<BATCH, 128, DEC_SMEM, stream>>>(demands, caps, Wp2, bp2,
                                                 attF, attB, out);
}
